// my_GraphConvolution1_33655363732151
// MI455X (gfx1250) — matrix operand order repaired and checked by reading, not compiled, not run on hardware
//
#include <hip/hip_runtime.h>
#include <hip/hip_bf16.h>

// out = adj @ (input @ W1 + W2) + x @ W3 + bias
//   input [8192,512] f32, adj [8192,8192] f32, x [8192,512] f32,
//   weight [9216,512] f32 (W1 rows 0..511, W2 rows 512..8703, W3 rows 8704..9215),
//   bias [512] f32, out [8192,512] f32.
//
// Workspace layout:
//   Tt  bf16 [512][8192]  (transposed T = input@W1 + W2)    8 MB   @ ws + 0
//   Bt  f32  [512][8192]  (transposed  x@W3 + bias)        16 MB   @ ws + 8MB
//   Wt1 bf16 [512][512]   (transposed W1)                 512 KB   @ ws + 24MB
//   Wt3 bf16 [512][512]   (transposed W3)                 512 KB   @ ws + 24.5MB

typedef __attribute__((ext_vector_type(16))) __bf16        v16bf;
typedef __attribute__((ext_vector_type(8)))  __bf16        v8bf;
typedef __attribute__((ext_vector_type(8)))  float         v8f;
typedef __attribute__((ext_vector_type(4)))  float         v4f;
typedef __attribute__((ext_vector_type(4)))  unsigned int  v4u;
typedef __attribute__((ext_vector_type(4)))  int           v4i;

#if defined(__AMDGCN__) && __has_builtin(__builtin_amdgcn_global_load_async_to_lds_b128) && \
    __has_builtin(__builtin_amdgcn_s_wait_asynccnt)
#define USE_ASYNC_LDS 1
// src: global (AS1) pointer to 16B; dst: LDS (AS3) pointer to 16B; imm offset, imm cpol
#define ASYNC_COPY_B128(gp, lp)                                              \
    __builtin_amdgcn_global_load_async_to_lds_b128(                          \
        (__attribute__((address_space(1))) v4i*)(gp),                        \
        (__attribute__((address_space(3))) v4i*)(lp), 0, 0)
#else
#define USE_ASYNC_LDS 0
#endif

union Frag {            // 16x32 (A) or 32x16 (B) bf16 fragment = 8 VGPRs
    v4u   u[2];
    v16bf bf;
};

__device__ __forceinline__ v4u pack8_bf16(v4f a, v4f b) {
    v8f f = __builtin_shufflevector(a, b, 0, 1, 2, 3, 4, 5, 6, 7);
    union { v8bf b8; v4u u; } r;
    r.b8 = __builtin_convertvector(f, v8bf);   // packed v_cvt_pk_bf16_f32
    return r.u;
}

__device__ __forceinline__ v8f wmma_bf16(const Frag& fa, const Frag& fb, v8f c) {
    return __builtin_amdgcn_wmma_f32_16x16x32_bf16(
        /*neg_a=*/false, fa.bf, /*neg_b=*/false, fb.bf,
        /*c_mod=*/(short)0, c, /*reuse_a=*/false, /*reuse_b=*/false);
}

// ---------------------------------------------------------------------------
// Kernel 0: one-shot transpose+convert of W1/W3 -> bf16 N-major Wt[n][k].
// Each thread emits one 16B chunk (8 bf16 along K). Tiny (1 MB written).
// ---------------------------------------------------------------------------
__global__ __launch_bounds__(256) void gcn_wt_convert_kernel(
    const float* __restrict__ weight, __bf16* __restrict__ Wt1, __bf16* __restrict__ Wt3)
{
    const int id = blockIdx.x * 256 + threadIdx.x;   // 512*512/8 = 32768 chunks
    const int n  = id >> 6;                          // 0..511
    const int kc = id & 63;                          // 16B chunk along K
    const float* src = weight + (blockIdx.y ? (size_t)(512 + 8192) * 512 : 0);
    __bf16*      dst = blockIdx.y ? Wt3 : Wt1;
    union { __bf16 h[8]; v4u u; } p;
#pragma unroll
    for (int j = 0; j < 8; ++j)
        p.h[j] = (__bf16)src[(size_t)(kc * 8 + j) * 512 + n];
    *(v4u*)(dst + (size_t)n * 512 + kc * 8) = p.u;
}

// ---------------------------------------------------------------------------
// Kernel 1: prep.  z==0:  Tt[n][m] = bf16( (input@W1)[m][n] + W2[m][n] )
//                  z==1:  Bt[n][m] = (x@W3)[m][n] + bias[n]
// Tile 64(M) x 128(N) x 32(K); B operand staged from pre-transposed Wt.
// ---------------------------------------------------------------------------
__global__ __launch_bounds__(256) void gcn_prep_kernel(
    const float* __restrict__ input, const float* __restrict__ x,
    const float* __restrict__ weight, const float* __restrict__ bias,
    const __bf16* __restrict__ Wt1, const __bf16* __restrict__ Wt3,
    __bf16* __restrict__ Tt, float* __restrict__ Bt)
{
    __shared__ v4u sA[64 * 5];    // 64 rows x 32 bf16, stride 80B (5 x 16B chunks)
    __shared__ v4u sB[128 * 5];   // 128 N-rows x 32 bf16 (N-major), stride 80B

    const int mode = blockIdx.z;
    const float*  A  = mode ? x : input;     // [8192,512] f32
    const __bf16* Wt = mode ? Wt3 : Wt1;     // [512,512] bf16, N-major

    const int tid  = threadIdx.x;
    const int lane = tid & 31;
    const int wave = tid >> 5;
    const int wr   = wave & 3;      // 16-row strip
    const int wc   = wave >> 2;     // 64-col half
    const int l15  = lane & 15;
    const int h    = lane >> 4;
    const int m0   = blockIdx.x * 64;
    const int n0   = blockIdx.y * 128;

    const int rowA = tid >> 2, cA = tid & 3;

    v8f acc[4] = {};
    for (int kk = 0; kk < 512; kk += 32) {
        { // A tile: 64x32 f32 -> bf16
            const float* gp = A + (size_t)(m0 + rowA) * 512 + kk + cA * 8;
            sA[rowA * 5 + cA] = pack8_bf16(*(const v4f*)gp, *(const v4f*)(gp + 4));
        }
#pragma unroll
        for (int i = 0; i < 2; ++i) { // B tile: 128 n-rows x 64B, pure 16B copies
            const int id  = tid + 256 * i;
            const int row = id >> 2, c = id & 3;
            const __bf16* gp = Wt + (size_t)(n0 + row) * 512 + kk + c * 8;
#if USE_ASYNC_LDS
            ASYNC_COPY_B128(gp, &sB[row * 5 + c]);
#else
            sB[row * 5 + c] = *(const v4u*)gp;
#endif
        }
#if USE_ASYNC_LDS
        __builtin_amdgcn_s_wait_asynccnt(0);
#endif
        __syncthreads();

        Frag fa;
        const int mrow = wr * 16 + l15;
        fa.u[0] = sA[mrow * 5 + h];
        fa.u[1] = sA[mrow * 5 + h + 2];
#pragma unroll
        for (int t = 0; t < 4; ++t) {
            Frag fb;
            const int nrow = wc * 64 + t * 16 + l15;
            fb.u[0] = sB[nrow * 5 + h];
            fb.u[1] = sB[nrow * 5 + h + 2];
            acc[t] = wmma_bf16(fa, fb, acc[t]);
        }
        __syncthreads();
    }

#pragma unroll
    for (int t = 0; t < 4; ++t) {
        const int n  = n0 + wc * 64 + t * 16 + l15;
        const int mb = m0 + wr * 16 + h * 8;
        if (mode == 0) {
            v4f a0, a1;
#pragma unroll
            for (int v = 0; v < 4; ++v) {
                a0[v] = acc[t][v]     + weight[(size_t)(512 + mb + v) * 512 + n];
                a1[v] = acc[t][4 + v] + weight[(size_t)(512 + mb + 4 + v) * 512 + n];
            }
            *(v4u*)(Tt + (size_t)n * 8192 + mb) = pack8_bf16(a0, a1);
        } else {
            const float bs = bias[n];
            v4f lo, hi;
#pragma unroll
            for (int v = 0; v < 4; ++v) { lo[v] = acc[t][v] + bs; hi[v] = acc[t][4 + v] + bs; }
            float* dst = Bt + (size_t)n * 8192 + mb;
            *(v4f*)dst       = lo;
            *(v4f*)(dst + 4) = hi;
        }
    }
}

// ---------------------------------------------------------------------------
// Kernel 2: out[m][n] = sum_k adj[m][k] * Tt[n][k] + Bt[n][m]
// Tile 64(M) x 256(N) x 32(K); double-buffered LDS software pipeline.
// A staged via VGPRs (f32->bf16 convert); B via async global->LDS when present.
// ---------------------------------------------------------------------------
__global__ __launch_bounds__(256) void gcn_main_kernel(
    const float* __restrict__ adj, const __bf16* __restrict__ Tt,
    const float* __restrict__ Bt, float* __restrict__ out)
{
    __shared__ v4u sA[2][64 * 5];     // 2 x 5 KB
    __shared__ v4u sB[2][256 * 5];    // 2 x 20 KB

    const int tid  = threadIdx.x;
    const int lane = tid & 31;
    const int wave = tid >> 5;
    const int wr   = wave & 3;        // 16-row strip
    const int wc   = wave >> 2;       // 128-column half
    const int l15  = lane & 15;
    const int h    = lane >> 4;
    const int m0   = blockIdx.x * 64;
    const int n0   = blockIdx.y * 256;

    const int rowA = tid >> 2, cA = tid & 3;
    const float*  gA = adj + (size_t)(m0 + rowA) * 8192 + cA * 8;  // + kk
    const int dstA = rowA * 5 + cA;

    constexpr int NSTEP = 8192 / 32;
    v8f acc[8] = {};

    // ---- prologue: stage tile 0 into buffer 0 ----
    {
        v4f a0 = *(const v4f*)gA, a1 = *(const v4f*)(gA + 4);
#pragma unroll
        for (int i = 0; i < 4; ++i) {
            const int id  = tid + 256 * i;
            const int row = id >> 2, c = id & 3;
            const __bf16* gp = Tt + (size_t)(n0 + row) * 8192 + c * 8;
#if USE_ASYNC_LDS
            ASYNC_COPY_B128(gp, &sB[0][row * 5 + c]);
#else
            sB[0][row * 5 + c] = *(const v4u*)gp;
#endif
        }
        sA[0][dstA] = pack8_bf16(a0, a1);
#if USE_ASYNC_LDS
        __builtin_amdgcn_s_wait_asynccnt(0);
#endif
        __syncthreads();
    }

    // ---- steady state: one barrier per K-step ----
    for (int s = 0; s < NSTEP; ++s) {
        const int cur = s & 1, nxt = cur ^ 1;
        const bool has_next = (s + 1) < NSTEP;
        const int kk_next = (s + 1) * 32;

        v4f a0, a1;
#if !USE_ASYNC_LDS
        v4u breg[4];
#endif
        if (has_next) {
            a0 = *(const v4f*)(gA + kk_next);
            a1 = *(const v4f*)(gA + kk_next + 4);
#pragma unroll
            for (int i = 0; i < 4; ++i) {
                const int id  = tid + 256 * i;
                const int row = id >> 2, c = id & 3;
                const __bf16* gp = Tt + (size_t)(n0 + row) * 8192 + kk_next + c * 8;
#if USE_ASYNC_LDS
                ASYNC_COPY_B128(gp, &sB[nxt][row * 5 + c]);
#else
                breg[i] = *(const v4u*)gp;
#endif
            }
        }

        // compute on current buffer (overlaps with in-flight loads)
        Frag fa;
        const int mrow = wr * 16 + l15;
        fa.u[0] = sA[cur][mrow * 5 + h];
        fa.u[1] = sA[cur][mrow * 5 + h + 2];
#pragma unroll
        for (int t = 0; t < 8; ++t) {
            Frag fb;
            const int nrow = wc * 128 + t * 16 + l15;
            fb.u[0] = sB[cur][nrow * 5 + h];
            fb.u[1] = sB[cur][nrow * 5 + h + 2];
            acc[t] = wmma_bf16(fa, fb, acc[t]);
        }

        if (has_next) {
            sA[nxt][dstA] = pack8_bf16(a0, a1);
#if !USE_ASYNC_LDS
#pragma unroll
            for (int i = 0; i < 4; ++i) {
                const int id  = tid + 256 * i;
                const int row = id >> 2, c = id & 3;
                sB[nxt][row * 5 + c] = breg[i];
            }
#endif
#if USE_ASYNC_LDS
            __builtin_amdgcn_s_wait_asynccnt(0);
#endif
        }
        __syncthreads();
    }

    // ---- epilogue: add transposed base, scatter f32 rows (coalesced per half-wave) ----
#pragma unroll
    for (int t = 0; t < 8; ++t) {
        const int n  = n0 + wc * 128 + t * 16 + l15;
        const int mb = m0 + wr * 16 + h * 8;
        const float* bp = Bt + (size_t)n * 8192 + mb;
        v4f b0 = *(const v4f*)bp;
        v4f b1 = *(const v4f*)(bp + 4);
#pragma unroll
        for (int v = 0; v < 4; ++v)
            out[(size_t)(mb + v) * 512 + n] = acc[t][v] + b0[v];
#pragma unroll
        for (int v = 0; v < 4; ++v)
            out[(size_t)(mb + 4 + v) * 512 + n] = acc[t][4 + v] + b1[v];
    }
}

extern "C" void kernel_launch(void* const* d_in, const int* in_sizes, int n_in,
                              void* d_out, int out_size, void* d_ws, size_t ws_size,
                              hipStream_t stream) {
    const float* input  = (const float*)d_in[0];   // [8192, 512]
    const float* adj    = (const float*)d_in[1];   // [8192, 8192]
    const float* x      = (const float*)d_in[2];   // [8192, 512]
    const float* weight = (const float*)d_in[3];   // [9216, 512]
    const float* bias   = (const float*)d_in[4];   // [512]
    float* out = (float*)d_out;                    // [8192, 512]

    char* ws = (char*)d_ws;
    __bf16* Tt  = (__bf16*)ws;                                  // 8 MB
    float*  Bt  = (float*)(ws + (size_t)512 * 8192 * 2);        // 16 MB
    __bf16* Wt1 = (__bf16*)(ws + (size_t)24 * 1024 * 1024);     // 512 KB
    __bf16* Wt3 = (__bf16*)(ws + (size_t)24 * 1024 * 1024 + 512 * 1024);

    dim3 blk(256, 1, 1);

    dim3 g0(512 * 512 / 8 / 256, 2, 1);   // Wt convert: 128 x 2
    gcn_wt_convert_kernel<<<g0, blk, 0, stream>>>(weight, Wt1, Wt3);

    dim3 g1(8192 / 64, 512 / 128, 2);     // prep: 128 x 4 tiles, z = mode
    gcn_prep_kernel<<<g1, blk, 0, stream>>>(input, x, weight, bias, Wt1, Wt3, Tt, Bt);

    dim3 g2(8192 / 64, 512 / 256, 1);     // main: 128 x 2 tiles
    gcn_main_kernel<<<g2, blk, 0, stream>>>(adj, Tt, Bt, out);
}
